// VehicleDetection_74208444940322
// MI455X (gfx1250) — matrix operand order repaired and checked by reading, not compiled, not run on hardware
//
#include <hip/hip_runtime.h>

typedef __attribute__((ext_vector_type(16))) _Float16 v16h;
typedef __attribute__((ext_vector_type(8)))  _Float16 v8h;
typedef __attribute__((ext_vector_type(8)))  float    v8f;

#define LRELU(v, s) ((v) >= 0.0f ? (v) : (v) * (s))
#define CAT16(x, y) __builtin_shufflevector(x, y, 0, 1, 2, 3, 4, 5, 6, 7, 8, 9, 10, 11, 12, 13, 14, 15)
#define WMMA(a, b, c) __builtin_amdgcn_wmma_f32_16x16x32_f16(false, a, false, b, (short)0, c, false, false)

__device__ inline v16h ldA(const _Float16* p, int hi)
{
    return CAT16(*(const v8h*)(p + 8 * hi), *(const v8h*)(p + 16 + 8 * hi));
}
__device__ inline v16h ldB(const _Float16* p, int hi)
{
    return CAT16(*(const v8h*)(p + 8 * hi), *(const v8h*)(p + 16 + 8 * hi));
}
__device__ inline void store_nlc(v8f av, _Float16* __restrict__ Yh,
                                 const _Float16* __restrict__ Resh,
                                 size_t off, float slope)
{
    v8h rv = {};
    if (Resh) rv = *(const v8h*)(Resh + off);
    v8h outv;
#pragma unroll
    for (int r = 0; r < 8; ++r) {
        float v = LRELU(av[r], slope);
        if (Resh) v += (float)rv[r];
        outv[r] = (_Float16)v;
    }
    *(v8h*)(Yh + off) = outv;
}

// ---------------------------------------------------------------------------
// Conv1d on f16 [B, L, C] activations, weights Wt[co][dk*Ci+ci] f16.
// 2x2 WMMA tiles per wave: 32 out-channels x 32 positions.
// dk-outer / ci-inner loop: boundary clamp + row-pointer select hoisted per
// tap; inner loop is pure b128 loads + 4 independent WMMAs.
// ---------------------------------------------------------------------------
template <int Ci, int Kw>
__global__ void conv_nlc_wmma(const _Float16* __restrict__ Xh,
                              const _Float16* __restrict__ Wt,
                              const float* __restrict__ Bias,
                              const _Float16* __restrict__ Resh,
                              _Float16* __restrict__ Yh,
                              float* __restrict__ Yf,
                              const _Float16* __restrict__ Zrow,
                              int Co, int L, float slope)
{
    constexpr int PAD  = (Kw - 1) / 2;
    constexpr int Ktot = Ci * Kw;
    const int lane = threadIdx.x & 31;
    const int wid  = threadIdx.x >> 5;
    const int lo   = lane & 15, hi = lane >> 4;
    const int l_base  = (blockIdx.x * (blockDim.x >> 5) + wid) * 32;
    const int co_base = blockIdx.y * 32;
    const int bz      = blockIdx.z;
    if (l_base >= L) return;

    const int n0 = l_base + lo;
    const int n1 = n0 + 16;

    v8f acc00, acc01, acc10, acc11;
#pragma unroll
    for (int r = 0; r < 8; ++r) {
        float bA = Bias[co_base + r + 8 * hi];
        float bB = Bias[co_base + 16 + r + 8 * hi];
        acc00[r] = bA; acc01[r] = bA;
        acc10[r] = bB; acc11[r] = bB;
    }

    const _Float16* W0 = Wt + (size_t)(co_base + lo) * Ktot;
    const _Float16* W1 = W0 + (size_t)16 * Ktot;
    const _Float16* Xb = Xh + (size_t)bz * L * Ci;

#pragma unroll
    for (int dk = 0; dk < Kw; ++dk) {
        const int  p0 = n0 + dk - PAD;
        const int  p1 = n1 + dk - PAD;
        const bool i0 = (p0 >= 0) && (p0 < L);
        const bool i1 = (p1 >= 0) && (p1 < L);
        const int  q0 = p0 < 0 ? 0 : (p0 >= L ? L - 1 : p0);
        const int  q1 = p1 < 0 ? 0 : (p1 >= L ? L - 1 : p1);
        const _Float16* s0 = i0 ? (Xb + (size_t)q0 * Ci) : Zrow;
        const _Float16* s1 = i1 ? (Xb + (size_t)q1 * Ci) : Zrow;
        const _Float16* wk0 = W0 + dk * Ci;
        const _Float16* wk1 = W1 + dk * Ci;
#pragma unroll
        for (int c0 = 0; c0 < Ci; c0 += 32) {
            const v16h a0 = ldA(wk0 + c0, hi);
            const v16h a1 = ldA(wk1 + c0, hi);
            const v16h b0 = ldB(s0 + c0, hi);
            const v16h b1 = ldB(s1 + c0, hi);
            acc00 = WMMA(a0, b0, acc00);
            acc01 = WMMA(a0, b1, acc01);
            acc10 = WMMA(a1, b0, acc10);
            acc11 = WMMA(a1, b1, acc11);
        }
    }

    if (Yf) {                                        // f32 NCL store (conv5)
#pragma unroll
        for (int r = 0; r < 8; ++r) {
            int m0 = co_base + r + 8 * hi;
            int m1 = m0 + 16;
            Yf[((size_t)bz * Co + m0) * L + n0] = LRELU(acc00[r], slope);
            Yf[((size_t)bz * Co + m0) * L + n1] = LRELU(acc01[r], slope);
            Yf[((size_t)bz * Co + m1) * L + n0] = LRELU(acc10[r], slope);
            Yf[((size_t)bz * Co + m1) * L + n1] = LRELU(acc11[r], slope);
        }
    } else {                                         // packed f16 NLC stores
        size_t row0 = ((size_t)bz * L + n0) * Co;
        size_t row1 = ((size_t)bz * L + n1) * Co;
        store_nlc(acc00, Yh, Resh, row0 + co_base + 8 * hi, slope);
        store_nlc(acc01, Yh, Resh, row1 + co_base + 8 * hi, slope);
        store_nlc(acc10, Yh, Resh, row0 + co_base + 16 + 8 * hi, slope);
        store_nlc(acc11, Yh, Resh, row1 + co_base + 16 + 8 * hi, slope);
    }
}

// ---------------------------------------------------------------------------
// ROI conv (C=128, L=5, k=3, pad=1) on f16 [item][5][128]; 32 items x 32
// out-channels per wave; fixed output position j = blockIdx.z.
// ---------------------------------------------------------------------------
__global__ void conv_roi_wmma(const _Float16* __restrict__ Xh,
                              const _Float16* __restrict__ Wt,
                              const float* __restrict__ Bias,
                              _Float16* __restrict__ Yh,
                              const _Float16* __restrict__ Zrow)
{
    const int lane = threadIdx.x & 31;
    const int wid  = threadIdx.x >> 5;
    const int lo   = lane & 15, hi = lane >> 4;
    const int it_base = (blockIdx.x * (blockDim.x >> 5) + wid) * 32;
    const int co_base = blockIdx.y * 32;
    const int j       = blockIdx.z;

    v8f acc00, acc01, acc10, acc11;
#pragma unroll
    for (int r = 0; r < 8; ++r) {
        float bA = Bias[co_base + r + 8 * hi];
        float bB = Bias[co_base + 16 + r + 8 * hi];
        acc00[r] = bA; acc01[r] = bA;
        acc10[r] = bB; acc11[r] = bB;
    }

    const _Float16* W0 = Wt + (size_t)(co_base + lo) * 384;
    const _Float16* W1 = W0 + (size_t)16 * 384;
    const _Float16* Xi0 = Xh + (size_t)(it_base + lo) * 640;
    const _Float16* Xi1 = Xi0 + (size_t)16 * 640;

#pragma unroll
    for (int dk = 0; dk < 3; ++dk) {
        const int  pos  = j + dk - 1;                // wave-uniform
        const bool inb  = (pos >= 0) && (pos < 5);
        const int  posc = pos < 0 ? 0 : (pos > 4 ? 4 : pos);
        const _Float16* s0 = inb ? (Xi0 + posc * 128) : Zrow;
        const _Float16* s1 = inb ? (Xi1 + posc * 128) : Zrow;
        const _Float16* wk0 = W0 + dk * 128;
        const _Float16* wk1 = W1 + dk * 128;
#pragma unroll
        for (int c0 = 0; c0 < 128; c0 += 32) {
            const v16h a0 = ldA(wk0 + c0, hi);
            const v16h a1 = ldA(wk1 + c0, hi);
            const v16h b0 = ldB(s0 + c0, hi);
            const v16h b1 = ldB(s1 + c0, hi);
            acc00 = WMMA(a0, b0, acc00);
            acc01 = WMMA(a0, b1, acc01);
            acc10 = WMMA(a1, b0, acc10);
            acc11 = WMMA(a1, b1, acc11);
        }
    }

    size_t r0 = ((size_t)(it_base + lo) * 5 + j) * 128;
    size_t r1 = ((size_t)(it_base + 16 + lo) * 5 + j) * 128;
    v8h o00, o01, o10, o11;
#pragma unroll
    for (int r = 0; r < 8; ++r) {
        o00[r] = (_Float16)acc00[r];
        o01[r] = (_Float16)acc01[r];
        o10[r] = (_Float16)acc10[r];
        o11[r] = (_Float16)acc11[r];
    }
    *(v8h*)(Yh + r0 + co_base + 8 * hi)      = o00;
    *(v8h*)(Yh + r1 + co_base + 8 * hi)      = o01;
    *(v8h*)(Yh + r0 + co_base + 16 + 8 * hi) = o10;
    *(v8h*)(Yh + r1 + co_base + 16 + 8 * hi) = o11;
}

// ---------------------------------------------------------------------------
// FC: Y[item][n] = lrelu(X[item][:K] . Wt[n][:K] + b[n]); 32 items x 32 feats
// per wave, all-f16 operands, contiguous b128 fragment loads.
// ---------------------------------------------------------------------------
__global__ void fc_lrelu_wmma(const _Float16* __restrict__ Xh,
                              const _Float16* __restrict__ Wt,
                              const float* __restrict__ Bias,
                              float* __restrict__ Y,
                              int K, int Nout, float slope)
{
    const int lane = threadIdx.x & 31;
    const int wid  = threadIdx.x >> 5;
    const int lo   = lane & 15, hi = lane >> 4;
    const int it_base = (blockIdx.x * (blockDim.x >> 5) + wid) * 32;
    const int n_base  = blockIdx.y * 32;

    v8f acc00, acc01, acc10, acc11;
    const float bA = Bias[n_base + lo];
    const float bB = Bias[n_base + 16 + lo];
#pragma unroll
    for (int r = 0; r < 8; ++r) {
        acc00[r] = bA; acc01[r] = bB;
        acc10[r] = bA; acc11[r] = bB;
    }

    const _Float16* A0 = Xh + (size_t)(it_base + lo) * K;
    const _Float16* A1 = A0 + (size_t)16 * K;
    const _Float16* B0 = Wt + (size_t)(n_base + lo) * K;
    const _Float16* B1 = B0 + (size_t)16 * K;

    for (int kt = 0; kt < K; kt += 32) {
        const v16h a0 = ldA(A0 + kt, hi);
        const v16h a1 = ldA(A1 + kt, hi);
        const v16h b0 = ldB(B0 + kt, hi);
        const v16h b1 = ldB(B1 + kt, hi);
        acc00 = WMMA(a0, b0, acc00);
        acc01 = WMMA(a0, b1, acc01);
        acc10 = WMMA(a1, b0, acc10);
        acc11 = WMMA(a1, b1, acc11);
    }
#pragma unroll
    for (int r = 0; r < 8; ++r) {
        int m0 = it_base + r + 8 * hi;
        int m1 = m0 + 16;
        Y[(size_t)m0 * Nout + n_base + lo]      = LRELU(acc00[r], slope);
        Y[(size_t)m0 * Nout + n_base + 16 + lo] = LRELU(acc01[r], slope);
        Y[(size_t)m1 * Nout + n_base + lo]      = LRELU(acc10[r], slope);
        Y[(size_t)m1 * Nout + n_base + 16 + lo] = LRELU(acc11[r], slope);
    }
}

// ---------------------------------------------------------------------------
// Glue / transform kernels
// ---------------------------------------------------------------------------
__global__ void zero_f16(_Float16* p, int n)
{
    int i = blockIdx.x * blockDim.x + threadIdx.x;
    if (i < n) p[i] = (_Float16)0.0f;
}

__global__ void eff12(const float* c1w, const float* c1b, const float* c2w,
                      const float* c2b, float* We, float* be)
{
    int i = blockIdx.x * blockDim.x + threadIdx.x;
    if (i < 384) {
        int o = i / 6, j = i - o * 6;
        float s = 0.0f;
        for (int m = 0; m < 24; ++m) s += c2w[o * 24 + m] * c1w[m * 6 + j];
        We[i] = s;
    } else if (i < 448) {
        int o = i - 384;
        float s = c2b[o];
        for (int m = 0; m < 24; ++m) s += c2w[o * 24 + m] * c1b[m];
        be[o] = s;
    }
}

__global__ void conv12_fused(const float* __restrict__ X, const float* __restrict__ We,
                             const float* __restrict__ be, _Float16* __restrict__ Yh, int L)
{
    int i = blockIdx.x * blockDim.x + threadIdx.x;
    if (i >= 32 * L * 8) return;
    int g   = i & 7;
    int pos = (i >> 3) % L;
    int b   = i / (L * 8);
    float x[6];
#pragma unroll
    for (int j = 0; j < 6; ++j) x[j] = X[((size_t)b * 6 + j) * L + pos];
    v8h out;
#pragma unroll
    for (int r = 0; r < 8; ++r) {
        int c = g * 8 + r;
        float s = be[c];
#pragma unroll
        for (int j = 0; j < 6; ++j) s += x[j] * We[c * 6 + j];
        out[r] = (_Float16)s;
    }
    *(v8h*)(Yh + ((size_t)b * L + pos) * 64 + g * 8) = out;
}

__global__ void wt_conv(const float* __restrict__ W, _Float16* __restrict__ Wt,
                        int Co, int Ci, int Kw)
{
    int i = blockIdx.x * blockDim.x + threadIdx.x;
    if (i >= Co * Ci * Kw) return;
    int co  = i / (Ci * Kw);
    int rem = i - co * (Ci * Kw);
    int ci  = rem / Kw;
    int dk  = rem - ci * Kw;
    Wt[(size_t)co * Ci * Kw + dk * Ci + ci] = (_Float16)W[i];
}

__global__ void wt_fc(const float* __restrict__ W, _Float16* __restrict__ Wt)
{
    int i = blockIdx.x * blockDim.x + threadIdx.x;
    if (i >= 128 * 640) return;
    int nrow = i / 640, k = i - nrow * 640;
    int c = k / 5, j = k - c * 5;
    Wt[(size_t)nrow * 640 + j * 128 + c] = (_Float16)W[i];
}

__global__ void avgpool2_h(const _Float16* __restrict__ X, _Float16* __restrict__ Y,
                           int total, int C)
{
    int i = blockIdx.x * blockDim.x + threadIdx.x;
    if (i >= total) return;
    int c   = i % C;
    int in0 = 2 * i - c;
    Y[i] = (_Float16)(0.5f * ((float)X[in0] + (float)X[in0 + C]));
}

__global__ void row_cumsum(const float* __restrict__ X, float* __restrict__ cs,
                           int rows, int F)
{
    int r = blockIdx.x * blockDim.x + threadIdx.x;
    if (r >= rows) return;
    const float* x = X + (size_t)r * F;
    float* c = cs + (size_t)r * (F + 1);
    float s = 0.0f;
    c[0] = 0.0f;
    for (int f = 0; f < F; ++f) { s += x[f]; c[f + 1] = s; }
}

__global__ void roi_gather(const float* __restrict__ cs, const float* __restrict__ P,
                           _Float16* __restrict__ roi, int Bn, int C, int F, int Nprop)
{
    int i = blockIdx.x * blockDim.x + threadIdx.x;
    if (i >= Nprop * Bn * C * 5) return;
    int j = i % 5;
    int c = (i / 5) % C;
    int b = (i / (5 * C)) % Bn;
    int n = i / (5 * C * Bn);
    float p0 = P[n * 5 + 0], p1 = P[n * 5 + 1], p2 = P[n * 5 + 2];
    int s  = (int)floorf(p0 / p2 * (float)F);
    int e  = (int)floorf(p1 / p2 * (float)F) + 1;
    int Lp = ((e + 1) < F ? (e + 1) : F) - s;
    int a  = (j * Lp) / 5;
    int bb = ((j + 1) * Lp + 4) / 5;
    const float* row = cs + (size_t)(b * C + c) * (F + 1);
    float v = (row[s + bb] - row[s + a]) / (float)(bb - a);
    roi[((size_t)(n * Bn + b) * 5 + j) * C + c] = (_Float16)v;
}

__global__ void head_precompute(const float* l2w, const float* l2b, const float* l3w, const float* l3b,
                                const float* c2w, const float* c2b, const float* c3w, const float* c3b,
                                float* W23L, float* b23L, float* W23C, float* b23C)
{
    int i = blockIdx.x * blockDim.x + threadIdx.x;
    if (i < 512) {
        int o = i >> 7, k = i & 127;
        float s = 0.0f;
        for (int m = 0; m < 32; ++m) s += l3w[o * 32 + m] * l2w[m * 128 + k];
        W23L[i] = s;
    } else if (i < 516) {
        int o = i - 512;
        float s = l3b[o];
        for (int m = 0; m < 32; ++m) s += l3w[o * 32 + m] * l2b[m];
        b23L[o] = s;
    } else if (i < 644) {
        int k = i - 516;
        float s = 0.0f;
        for (int m = 0; m < 32; ++m) s += c3w[m] * c2w[m * 128 + k];
        W23C[k] = s;
    } else if (i == 644) {
        float s = c3b[0];
        for (int m = 0; m < 32; ++m) s += c3w[m] * c2b[m];
        b23C[0] = s;
    }
}

__global__ void head_final(const float* __restrict__ tloc, const float* __restrict__ tcls,
                           const float* __restrict__ W23L, const float* __restrict__ b23L,
                           const float* __restrict__ W23C, const float* __restrict__ b23C,
                           float* __restrict__ out, int M)
{
    int i = blockIdx.x * blockDim.x + threadIdx.x;
    if (i >= M * 5) return;
    int m = i / 5, o = i - m * 5;
    if (o < 4) {
        float s = b23L[o];
        const float* t = tloc + (size_t)m * 128;
        for (int k = 0; k < 128; ++k) s += t[k] * W23L[o * 128 + k];
        out[m * 4 + o] = s;
    } else {
        float s = b23C[0];
        const float* t = tcls + (size_t)m * 128;
        for (int k = 0; k < 128; ++k) s += t[k] * W23C[k];
        out[(size_t)M * 4 + m] = s;
    }
}

// ---------------------------------------------------------------------------
extern "C" void kernel_launch(void* const* d_in, const int* in_sizes, int n_in,
                              void* d_out, int out_size, void* d_ws, size_t ws_size,
                              hipStream_t stream)
{
    auto F = [&](int i) { return (const float*)d_in[i]; };
    const float* input = F(0);
    const float* props = F(1);
    const float *c1w = F(2),  *c1b = F(3),  *c2w = F(4),  *c2b = F(5);
    const float *b1w1 = F(6),  *b1b1 = F(7),  *b1w2 = F(8),  *b1b2 = F(9),  *b1w3 = F(10), *b1b3 = F(11);
    const float *b2w1 = F(12), *b2b1 = F(13), *b2w2 = F(14), *b2b2 = F(15), *b2w3 = F(16), *b2b3 = F(17);
    const float *crw = F(18), *crb = F(19);
    const float *b3w1 = F(20), *b3b1 = F(21), *b3w2 = F(22), *b3b2 = F(23), *b3w3 = F(24), *b3b3 = F(25);
    const float *b4w1 = F(26), *b4b1 = F(27), *b4w2 = F(28), *b4b2 = F(29), *b4w3 = F(30), *b4b3 = F(31);
    const float *c5w = F(32), *c5b = F(33);
    const float *h1w = F(34), *h1b = F(35), *h2w = F(36), *h2b = F(37), *h3w = F(38), *h3b = F(39);
    const float *l1w = F(40), *l1b = F(41), *l2w = F(42), *l2b = F(43), *l3w = F(44), *l3b = F(45);
    const float *q1w = F(46), *q1b = F(47), *q2w = F(48), *q2b = F(49), *q3w = F(50), *q3b = F(51);
    (void)n_in; (void)in_sizes; (void)out_size; (void)ws_size;

    char* base = (char*)d_ws;
    const size_t SZA = (size_t)17 * 1024 * 1024;            // f16 activation regions
    _Float16* A0 = (_Float16*)(base);
    _Float16* A1 = (_Float16*)(base + SZA);
    _Float16* A2 = (_Float16*)(base + 2 * SZA);
    float* FM = (float*)(base + 3 * SZA);                    // f32 NCL feature map
    float* CS = (float*)(base + 3 * SZA + (size_t)34 * 1024 * 1024);
    _Float16* WT = (_Float16*)(base + 3 * SZA + (size_t)68 * 1024 * 1024);
    float* SM = (float*)(base + 3 * SZA + (size_t)70 * 1024 * 1024);

    float* We = SM;            float* be = SM + 384;
    float* W23L = SM + 448;    float* b23L = SM + 960;
    float* W23C = SM + 964;    float* b23C = SM + 1092;
    _Float16* Zrow = (_Float16*)(SM + 1200);                 // 128 zero halves
    float* TLOC = FM;                                        // FM free after cumsum
    float* TCLS = FM + 524288;

    size_t wo = 0;
    auto walloc = [&](size_t n) { _Float16* p = WT + wo; wo += n; return p; };
    _Float16 *b1w1t = walloc(12288), *b1w2t = walloc(4096),  *b1w3t = walloc(12288);
    _Float16 *b2w1t = walloc(12288), *b2w2t = walloc(4096),  *b2w3t = walloc(12288);
    _Float16 *crwt  = walloc(8192);
    _Float16 *b3w1t = walloc(49152), *b3w2t = walloc(16384), *b3w3t = walloc(49152);
    _Float16 *b4w1t = walloc(49152), *b4w2t = walloc(16384), *b4w3t = walloc(49152);
    _Float16 *c5wt  = walloc(16384);
    _Float16 *h1wt  = walloc(49152), *h2wt = walloc(49152), *h3wt = walloc(49152);
    _Float16 *l1wt  = walloc(81920), *q1wt = walloc(81920);

    const dim3 blk(128);                                     // 4 wave32 per block
    auto cg = [](int L, int Co, int B) { return dim3(L / 128, Co / 32, B); };
    auto eg = [](int tot) { return dim3((tot + 255) / 256); };

    // ---- per-call weight prep (tiny) ----
    zero_f16<<<1, 128, 0, stream>>>(Zrow, 128);
    eff12<<<2, 256, 0, stream>>>(c1w, c1b, c2w, c2b, We, be);
    wt_conv<<<eg(12288), 256, 0, stream>>>(b1w1, b1w1t, 64, 64, 3);
    wt_conv<<<eg(4096),  256, 0, stream>>>(b1w2, b1w2t, 64, 64, 1);
    wt_conv<<<eg(12288), 256, 0, stream>>>(b1w3, b1w3t, 64, 64, 3);
    wt_conv<<<eg(12288), 256, 0, stream>>>(b2w1, b2w1t, 64, 64, 3);
    wt_conv<<<eg(4096),  256, 0, stream>>>(b2w2, b2w2t, 64, 64, 1);
    wt_conv<<<eg(12288), 256, 0, stream>>>(b2w3, b2w3t, 64, 64, 3);
    wt_conv<<<eg(8192),  256, 0, stream>>>(crw,  crwt, 128, 64, 1);
    wt_conv<<<eg(49152), 256, 0, stream>>>(b3w1, b3w1t, 128, 128, 3);
    wt_conv<<<eg(16384), 256, 0, stream>>>(b3w2, b3w2t, 128, 128, 1);
    wt_conv<<<eg(49152), 256, 0, stream>>>(b3w3, b3w3t, 128, 128, 3);
    wt_conv<<<eg(49152), 256, 0, stream>>>(b4w1, b4w1t, 128, 128, 3);
    wt_conv<<<eg(16384), 256, 0, stream>>>(b4w2, b4w2t, 128, 128, 1);
    wt_conv<<<eg(49152), 256, 0, stream>>>(b4w3, b4w3t, 128, 128, 3);
    wt_conv<<<eg(16384), 256, 0, stream>>>(c5w,  c5wt, 128, 128, 1);
    wt_conv<<<eg(49152), 256, 0, stream>>>(h1w,  h1wt, 128, 128, 3);
    wt_conv<<<eg(49152), 256, 0, stream>>>(h2w,  h2wt, 128, 128, 3);
    wt_conv<<<eg(49152), 256, 0, stream>>>(h3w,  h3wt, 128, 128, 3);
    wt_fc<<<eg(81920), 256, 0, stream>>>(l1w, l1wt);
    wt_fc<<<eg(81920), 256, 0, stream>>>(q1w, q1wt);

    // ---- backbone ----
    conv12_fused<<<eg(32 * 4096 * 8), 256, 0, stream>>>(input, We, be, A0, 4096);
    // blk1 (identity on 3rd conv, then residual)
    conv_nlc_wmma<64, 3><<<cg(4096, 64, 32), blk, 0, stream>>>(A0, b1w1t, b1b1, nullptr, A1, nullptr, Zrow, 64, 4096, 0.05f);
    conv_nlc_wmma<64, 1><<<cg(4096, 64, 32), blk, 0, stream>>>(A1, b1w2t, b1b2, nullptr, A2, nullptr, Zrow, 64, 4096, 0.05f);
    conv_nlc_wmma<64, 3><<<cg(4096, 64, 32), blk, 0, stream>>>(A2, b1w3t, b1b3, A0,      A1, nullptr, Zrow, 64, 4096, 1.0f);
    // blk2
    conv_nlc_wmma<64, 3><<<cg(4096, 64, 32), blk, 0, stream>>>(A1, b2w1t, b2b1, nullptr, A2, nullptr, Zrow, 64, 4096, 0.05f);
    conv_nlc_wmma<64, 1><<<cg(4096, 64, 32), blk, 0, stream>>>(A2, b2w2t, b2b2, nullptr, A0, nullptr, Zrow, 64, 4096, 0.05f);
    conv_nlc_wmma<64, 3><<<cg(4096, 64, 32), blk, 0, stream>>>(A0, b2w3t, b2b3, A1,      A2, nullptr, Zrow, 64, 4096, 1.0f);
    // pool then cross (k=1 linear commutes with avg-pool)
    avgpool2_h<<<eg(32 * 2048 * 64), 256, 0, stream>>>(A2, A0, 32 * 2048 * 64, 64);
    conv_nlc_wmma<64, 1><<<cg(2048, 128, 32), blk, 0, stream>>>(A0, crwt, crb, nullptr, A1, nullptr, Zrow, 128, 2048, 1.0f);
    // blk3 (lrelu 0.05 on 3rd conv, then residual)
    conv_nlc_wmma<128, 3><<<cg(2048, 128, 32), blk, 0, stream>>>(A1, b3w1t, b3b1, nullptr, A2, nullptr, Zrow, 128, 2048, 0.05f);
    conv_nlc_wmma<128, 1><<<cg(2048, 128, 32), blk, 0, stream>>>(A2, b3w2t, b3b2, nullptr, A0, nullptr, Zrow, 128, 2048, 0.05f);
    conv_nlc_wmma<128, 3><<<cg(2048, 128, 32), blk, 0, stream>>>(A0, b3w3t, b3b3, A1,      A2, nullptr, Zrow, 128, 2048, 0.05f);
    // blk4
    conv_nlc_wmma<128, 3><<<cg(2048, 128, 32), blk, 0, stream>>>(A2, b4w1t, b4b1, nullptr, A0, nullptr, Zrow, 128, 2048, 0.05f);
    conv_nlc_wmma<128, 1><<<cg(2048, 128, 32), blk, 0, stream>>>(A0, b4w2t, b4b2, nullptr, A1, nullptr, Zrow, 128, 2048, 0.05f);
    conv_nlc_wmma<128, 3><<<cg(2048, 128, 32), blk, 0, stream>>>(A1, b4w3t, b4b3, A2,      A0, nullptr, Zrow, 128, 2048, 0.05f);
    // conv5 -> f32 NCL feature map
    conv_nlc_wmma<128, 1><<<cg(2048, 128, 32), blk, 0, stream>>>(A0, c5wt, c5b, nullptr, nullptr, FM, Zrow, 128, 2048, 1.0f);

    // ---- ROI pooling ----
    row_cumsum<<<16, 256, 0, stream>>>(FM, CS, 32 * 128, 2048);
    roi_gather<<<eg(128 * 32 * 128 * 5), 256, 0, stream>>>(CS, props, A1, 32, 128, 2048, 128);

    // ---- ROI conv head (no activation) ----
    const dim3 rgrid(32, 4, 5);
    conv_roi_wmma<<<rgrid, blk, 0, stream>>>(A1, h1wt, h1b, A2, Zrow);
    conv_roi_wmma<<<rgrid, blk, 0, stream>>>(A2, h2wt, h2b, A0, Zrow);
    conv_roi_wmma<<<rgrid, blk, 0, stream>>>(A0, h3wt, h3b, A1, Zrow);

    // ---- FC heads ----
    const dim3 fgrid(32, 4);
    fc_lrelu_wmma<<<fgrid, blk, 0, stream>>>(A1, l1wt, l1b, TLOC, 640, 128, 0.1f);
    fc_lrelu_wmma<<<fgrid, blk, 0, stream>>>(A1, q1wt, q1b, TCLS, 640, 128, 0.1f);
    head_precompute<<<4, 256, 0, stream>>>(l2w, l2b, l3w, l3b, q2w, q2b, q3w, q3b,
                                           W23L, b23L, W23C, b23C);
    head_final<<<80, 256, 0, stream>>>(TLOC, TCLS, W23L, b23L, W23C, b23C,
                                       (float*)d_out, 4096);
}
